// EncoderLayer_30949534335352
// MI455X (gfx1250) — matrix operand order repaired and checked by reading, not compiled, not run on hardware
//
#include <hip/hip_runtime.h>
#include <hip/hip_bf16.h>

typedef __attribute__((ext_vector_type(16))) _Float16 v16h;
typedef __attribute__((ext_vector_type(8)))  _Float16 v8h;
typedef __attribute__((ext_vector_type(2)))  _Float16 v2h;
typedef __attribute__((ext_vector_type(8)))  float    v8f;

// ---------------------------------------------------------------------------
// Async byte copy of one 16B packet: global -> LDS, tracked on ASYNCcnt.
// Low 32 bits of a generic shared-aperture address are the LDS byte offset
// (ISA 10.2).
// ---------------------------------------------------------------------------
__device__ __forceinline__ void async_copy16(const void* gsrc, void* ldst) {
  unsigned           lds = (unsigned)(size_t)ldst;
  unsigned long long ga  = (unsigned long long)(size_t)gsrc;
  asm volatile("global_load_async_to_lds_b128 %0, %1, off"
               :: "v"(lds), "v"(ga) : "memory");
}
__device__ __forceinline__ void async_wait0() {
  asm volatile("s_wait_asynccnt 0x0" ::: "memory");
}

// ---------------------------------------------------------------------------
// Pipelined WMMA GEMM: C[M,N] = epilogue( scale * A[M,K] @ B )  (tile-exact)
//   TA/TB in {float,_Float16}: f16 A and f16 [N,K] B are async-copied LDS-direct;
//   the rest ride global loads -> cvt to f16 -> ds_store (b128; b32 pairs for [K,N] B).
//   B_TRANS=0: B is [K,N] row-major ; B_TRANS=1: B is [N,K] row-major (A@B^T)
// Block 256 threads / 8 waves; block tile 128x128; K-step 32; 2-deep LDS
// pipeline (stage tile i+1 while WMMAs consume tile i). Wave tile 32x64 ->
// 8 accumulators, 8 WMMAs per K-step.
// ---------------------------------------------------------------------------
template <typename TA, typename TB, typename TO,
          bool B_TRANS, bool BIAS, bool RES, bool RELU>
__global__ __launch_bounds__(256) void gemm_wmma(
    const TA* __restrict__ A, const TB* __restrict__ B,
    const float* __restrict__ bias, const float* __restrict__ res,
    TO* __restrict__ C, int M, int N, int K, float scale,
    long long strideA, long long strideB, long long strideC) {
  constexpr bool ASYNC_A = (sizeof(TA) == 2);
  constexpr bool ASYNC_B = (B_TRANS && sizeof(TB) == 2);
  constexpr bool HAS_ASYNC = ASYNC_A || ASYNC_B;

  const int tid  = threadIdx.x;
  const int wid  = tid >> 5;
  const int lane = tid & 31;
  const int g    = lane >> 4;   // half-wave select
  const int ln   = lane & 15;
  const int wave_m = wid & 3;
  const int wave_n = wid >> 2;
  const int mBase = blockIdx.x * 128;
  const int nBase = blockIdx.y * 128;

  A += (size_t)blockIdx.z * strideA;
  B += (size_t)blockIdx.z * strideB;
  C += (size_t)blockIdx.z * strideC;

  __shared__ _Float16 sA[2][128][32];  // [buf][m][k]
  __shared__ _Float16 sB[2][128][32];  // [buf][n][k]

  float aR[16];   // f32-A staging registers (dead code when TA==f16)
  float bR[16];   // f32-B / transposing-B staging registers

  // ---- issue global traffic for one K-tile (async goes straight to LDS) ----
  auto stage_issue = [&](int kk, int buf) {
    if constexpr (ASYNC_A) {
#pragma unroll
      for (int p = 0; p < 2; ++p) {
        const int c = tid + p * 256, row = c >> 2, h0 = (c & 3) * 8;
        async_copy16(A + (size_t)(mBase + row) * K + (kk + h0), &sA[buf][row][h0]);
      }
    } else {
      const float* ap = (const float*)A +
          (size_t)(mBase + (tid >> 1)) * K + (kk + (tid & 1) * 16);
#pragma unroll
      for (int e = 0; e < 16; ++e) aR[e] = ap[e];
    }
    if constexpr (B_TRANS) {
      if constexpr (ASYNC_B) {
#pragma unroll
        for (int p = 0; p < 2; ++p) {
          const int c = tid + p * 256, row = c >> 2, h0 = (c & 3) * 8;
          async_copy16(B + (size_t)(nBase + row) * K + (kk + h0), &sB[buf][row][h0]);
        }
      } else {
        const float* bp = (const float*)B +
            (size_t)(nBase + (tid >> 1)) * K + (kk + (tid & 1) * 16);
#pragma unroll
        for (int e = 0; e < 16; ++e) bR[e] = bp[e];
      }
    } else {  // B is [K,N]: 2 rows x 8 cols per thread
      const TB* b0p = B + (size_t)(kk + (tid >> 4) * 2) * N + (nBase + (tid & 15) * 8);
      const TB* b1p = b0p + N;
#pragma unroll
      for (int j = 0; j < 8; ++j) { bR[j] = (float)b0p[j]; bR[8 + j] = (float)b1p[j]; }
    }
  };

  // ---- convert staged registers and commit to LDS (non-async operands only) ----
  auto stage_commit = [&](int buf) {
    if constexpr (!ASYNC_A) {
      const int r = tid >> 1, c0 = (tid & 1) * 16;
      v8h p0, p1;
#pragma unroll
      for (int e = 0; e < 8; ++e) { p0[e] = (_Float16)aR[e]; p1[e] = (_Float16)aR[8 + e]; }
      *reinterpret_cast<v8h*>(&sA[buf][r][c0])     = p0;
      *reinterpret_cast<v8h*>(&sA[buf][r][c0 + 8]) = p1;
    }
    if constexpr (B_TRANS) {
      if constexpr (!ASYNC_B) {
        const int r = tid >> 1, c0 = (tid & 1) * 16;
        v8h p0, p1;
#pragma unroll
        for (int e = 0; e < 8; ++e) { p0[e] = (_Float16)bR[e]; p1[e] = (_Float16)bR[8 + e]; }
        *reinterpret_cast<v8h*>(&sB[buf][r][c0])     = p0;
        *reinterpret_cast<v8h*>(&sB[buf][r][c0 + 8]) = p1;
      }
    } else {
      const int n0 = (tid & 15) * 8, kk2 = (tid >> 4) * 2;
#pragma unroll
      for (int j = 0; j < 8; ++j) {
        v2h pk; pk[0] = (_Float16)bR[j]; pk[1] = (_Float16)bR[8 + j];
        *reinterpret_cast<v2h*>(&sB[buf][n0 + j][kk2]) = pk;
      }
    }
  };

  v8f acc[2][4] = {};

  // -------- prologue: tile 0 into buffer 0 --------
  stage_issue(0, 0);
  stage_commit(0);
  if constexpr (HAS_ASYNC) async_wait0();
  __syncthreads();

  const int nk = K >> 5;
  for (int i = 0; i < nk; ++i) {
    const int cur = i & 1, nxt = cur ^ 1;
    const bool more = (i + 1 < nk);
    if (more) stage_issue((i + 1) << 5, nxt);   // overlaps with WMMAs below

    // ---- fragments from buf[cur]: 2 A frags + 4 B frags (12 x ds_load_b128)
    v16h afrag[2], bfrag[4];
#pragma unroll
    for (int mt = 0; mt < 2; ++mt) {
      const _Float16* arow = &sA[cur][wave_m * 32 + mt * 16 + ln][0];
      v8h lo = *reinterpret_cast<const v8h*>(arow + g * 8);       // K = g*8..
      v8h hi = *reinterpret_cast<const v8h*>(arow + 16 + g * 8);  // K = 16+g*8..
#pragma unroll
      for (int e = 0; e < 8; ++e) { afrag[mt][e] = lo[e]; afrag[mt][e + 8] = hi[e]; }
    }
#pragma unroll
    for (int t = 0; t < 4; ++t) {
      const _Float16* brow = &sB[cur][wave_n * 64 + t * 16 + ln][0];
      v8h b0 = *reinterpret_cast<const v8h*>(brow + g * 8);
      v8h b1 = *reinterpret_cast<const v8h*>(brow + 16 + g * 8);
#pragma unroll
      for (int e = 0; e < 8; ++e) { bfrag[t][e] = b0[e]; bfrag[t][e + 8] = b1[e]; }
    }
    // ---- 8 back-to-back WMMAs ----
#pragma unroll
    for (int t = 0; t < 4; ++t)
#pragma unroll
      for (int mt = 0; mt < 2; ++mt)
        acc[mt][t] = __builtin_amdgcn_wmma_f32_16x16x32_f16(
            false, afrag[mt], false, bfrag[t], (short)0, acc[mt][t], false, false);

    if (more) stage_commit(nxt);        // cvt+ds_store after the math
    if constexpr (HAS_ASYNC) async_wait0();
    __syncthreads();
  }

  // ---- epilogue + store (D layout: VGPR r -> row r + 8*g, col = ln) ----
#pragma unroll
  for (int mt = 0; mt < 2; ++mt) {
#pragma unroll
    for (int t = 0; t < 4; ++t) {
      const int col = nBase + wave_n * 64 + t * 16 + ln;
      const float bv = BIAS ? bias[col] : 0.0f;
#pragma unroll
      for (int r = 0; r < 8; ++r) {
        const int row = mBase + wave_m * 32 + mt * 16 + g * 8 + r;
        float y = acc[mt][t][r] * scale;
        if (BIAS) y += bv;
        if (RES)  y += res[(size_t)row * N + col];
        if (RELU) y = fmaxf(y, 0.0f);
        C[(size_t)row * N + col] = (TO)y;
      }
    }
  }
}

// ---------------------------------------------------------------------------
// Masked row softmax: fp32 scores in, f16 attention out. Block per row (2048).
// ---------------------------------------------------------------------------
__global__ __launch_bounds__(256) void softmax_mask_rows(
    const float* __restrict__ S, const int* __restrict__ mask,
    _Float16* __restrict__ O, int n) {
  const size_t rowoff = (size_t)blockIdx.x * n;
  const float* row = S + rowoff;
  const int* mrow = mask + rowoff;
  _Float16* orow = O + rowoff;
  const int tid = threadIdx.x, lane = tid & 31, wid = tid >> 5;

  __shared__ float redm[8];
  __shared__ float reds[8];

  float v[8];
  float mx = -__builtin_inff();
#pragma unroll
  for (int j = 0; j < 8; ++j) {
    const int i = tid + j * 256;
    float x = (mrow[i] == 0) ? -__builtin_inff() : row[i];
    v[j] = x;
    mx = fmaxf(mx, x);
  }
#pragma unroll
  for (int o = 16; o > 0; o >>= 1) mx = fmaxf(mx, __shfl_xor(mx, o, 32));
  if (lane == 0) redm[wid] = mx;
  __syncthreads();
  mx = redm[0];
#pragma unroll
  for (int i = 1; i < 8; ++i) mx = fmaxf(mx, redm[i]);

  float sum = 0.0f;
#pragma unroll
  for (int j = 0; j < 8; ++j) { v[j] = __expf(v[j] - mx); sum += v[j]; }
#pragma unroll
  for (int o = 16; o > 0; o >>= 1) sum += __shfl_xor(sum, o, 32);
  if (lane == 0) reds[wid] = sum;
  __syncthreads();
  sum = reds[0];
#pragma unroll
  for (int i = 1; i < 8; ++i) sum += reds[i];

  const float inv = 1.0f / sum;
#pragma unroll
  for (int j = 0; j < 8; ++j) orow[tid + j * 256] = (_Float16)(v[j] * inv);
}

// ---------------------------------------------------------------------------
// LayerNorm, one block (256 threads) per row of length 1024.
// ---------------------------------------------------------------------------
__global__ __launch_bounds__(256) void layernorm_rows(
    const float* __restrict__ X, const float* __restrict__ gam,
    const float* __restrict__ bet, float* __restrict__ Y, int D) {
  const size_t rowoff = (size_t)blockIdx.x * D;
  const float* row = X + rowoff;
  const int tid = threadIdx.x, lane = tid & 31, wid = tid >> 5;

  __shared__ float red1[8];
  __shared__ float red2[8];

  float v[4], s = 0.0f, s2 = 0.0f;
#pragma unroll
  for (int j = 0; j < 4; ++j) {
    const int i = tid * 4 + j;
    float x = row[i];
    v[j] = x;
    s += x;
    s2 += x * x;
  }
#pragma unroll
  for (int o = 16; o > 0; o >>= 1) { s += __shfl_xor(s, o, 32); s2 += __shfl_xor(s2, o, 32); }
  if (lane == 0) { red1[wid] = s; red2[wid] = s2; }
  __syncthreads();
  s = red1[0]; s2 = red2[0];
#pragma unroll
  for (int i = 1; i < 8; ++i) { s += red1[i]; s2 += red2[i]; }

  const float invD = 1.0f / (float)D;
  const float mu = s * invD;
  const float var = s2 * invD - mu * mu;
  const float rstd = rsqrtf(var + 1e-5f);
#pragma unroll
  for (int j = 0; j < 4; ++j) {
    const int i = tid * 4 + j;
    Y[rowoff + i] = (v[j] - mu) * rstd * gam[i] + bet[i];
  }
}

// ---------------------------------------------------------------------------
extern "C" void kernel_launch(void* const* d_in, const int* in_sizes, int n_in,
                              void* d_out, int out_size, void* d_ws, size_t ws_size,
                              hipStream_t stream) {
  constexpr int Bb = 4, S = 2048, D = 1024, F = 4096;
  constexpr int M = Bb * S;                 // 8192 flattened rows
  const float SCALE = 0.03125f;             // 1/sqrt(1024)

  const float* x    = (const float*)d_in[0];
  const int*   mask = (const int*)  d_in[1];
  const float* wq = (const float*)d_in[2];  const float* bq = (const float*)d_in[3];
  const float* wk = (const float*)d_in[4];  const float* bk = (const float*)d_in[5];
  const float* wv = (const float*)d_in[6];  const float* bv = (const float*)d_in[7];
  const float* wo = (const float*)d_in[8];  const float* bo = (const float*)d_in[9];
  const float* w1 = (const float*)d_in[10]; const float* b1 = (const float*)d_in[11];
  const float* w2 = (const float*)d_in[12]; const float* b2 = (const float*)d_in[13];
  const float* g1 = (const float*)d_in[14]; const float* be1 = (const float*)d_in[15];
  const float* g2 = (const float*)d_in[16]; const float* be2 = (const float*)d_in[17];
  float* out = (float*)d_out;

  // ---- workspace layout (bytes), liveness-based reuse ----
  char* w = (char*)d_ws;
  const size_t SZ_MDh = (size_t)M * D * 2;       //  16.78 MB (f16 M x D)
  const size_t SZ_SSf = (size_t)Bb * S * S * 4;  //  67.11 MB (f32 scores)
  const size_t SZ_SSh = (size_t)Bb * S * S * 2;  //  33.55 MB (f16 attn)
  _Float16* Qh  = (_Float16*)(w);                                // f16 Q
  _Float16* Kh  = (_Float16*)(w + SZ_MDh);                       // f16 K
  _Float16* Vh  = (_Float16*)(w + 2 * SZ_MDh);                   // f16 V
  float*    Sb  = (float*)   (w + 3 * SZ_MDh);                   // f32 scores
  _Float16* Ah  = (_Float16*)(w + 3 * SZ_MDh + SZ_SSf);          // f16 attn
  _Float16* FFh = (_Float16*)(w + 3 * SZ_MDh + SZ_SSf + SZ_SSh); // f16 relu(ff)
  _Float16* CTXh = Qh;                           // Q dead after scores
  float* P1 = (float*)(w + 3 * SZ_MDh);          // scores dead after softmax
  float* H  = (float*)(w + 3 * SZ_MDh + SZ_SSf); // attn dead after ctx
  float* P2 = (float*)(w + SZ_MDh);              // K,V dead after ctx

  const dim3 blk(256);

  // 1) Q/K/V projections -> f16: [8192,1024] @ [1024,1024] + bias
  const dim3 gQKV(M / 128, D / 128, 1);
  gemm_wmma<float, float, _Float16, false, true, false, false>
      <<<gQKV, blk, 0, stream>>>(x, wq, bq, nullptr, Qh, M, D, D, 1.0f, 0, 0, 0);
  gemm_wmma<float, float, _Float16, false, true, false, false>
      <<<gQKV, blk, 0, stream>>>(x, wk, bk, nullptr, Kh, M, D, D, 1.0f, 0, 0, 0);
  gemm_wmma<float, float, _Float16, false, true, false, false>
      <<<gQKV, blk, 0, stream>>>(x, wv, bv, nullptr, Vh, M, D, D, 1.0f, 0, 0, 0);

  // 2) scores = Q @ K^T / sqrt(D) (f16 in via async-LDS pipeline, f32 out)
  gemm_wmma<_Float16, _Float16, float, true, false, false, false>
      <<<dim3(S / 128, S / 128, Bb), blk, 0, stream>>>(
          Qh, Kh, nullptr, nullptr, Sb, S, S, D, SCALE,
          (long long)S * D, (long long)S * D, (long long)S * S);

  // 3) masked softmax (f32 -> f16 attn)
  softmax_mask_rows<<<Bb * S, blk, 0, stream>>>(Sb, mask, Ah, S);

  // 4) ctx = attn @ V (f16 x f16 -> f16), batched
  gemm_wmma<_Float16, _Float16, _Float16, false, false, false, false>
      <<<dim3(S / 128, D / 128, Bb), blk, 0, stream>>>(
          Ah, Vh, nullptr, nullptr, CTXh, S, D, S, 1.0f,
          (long long)S * S, (long long)S * D, (long long)S * D);

  // 5) P1 = x + ctx @ wo + bo  (residual in epilogue, f32 out)
  gemm_wmma<_Float16, float, float, false, true, true, false>
      <<<dim3(M / 128, D / 128, 1), blk, 0, stream>>>(
          CTXh, wo, bo, x, P1, M, D, D, 1.0f, 0, 0, 0);

  // 6) h = LayerNorm1(P1)
  layernorm_rows<<<M, blk, 0, stream>>>(P1, g1, be1, H, D);

  // 7) ff = relu(h @ w1 + b1) -> f16
  gemm_wmma<float, float, _Float16, false, true, false, true>
      <<<dim3(M / 128, F / 128, 1), blk, 0, stream>>>(
          H, w1, b1, nullptr, FFh, M, F, D, 1.0f, 0, 0, 0);

  // 8) P2 = h + ff @ w2 + b2
  gemm_wmma<_Float16, float, float, false, true, true, false>
      <<<dim3(M / 128, D / 128, 1), blk, 0, stream>>>(
          FFh, w2, b2, H, P2, M, D, F, 1.0f, 0, 0, 0);

  // 9) out = LayerNorm2(P2)
  layernorm_rows<<<M, blk, 0, stream>>>(P2, g2, be2, out, D);
}
